// MultiScaleDynamicFusionGate_70738111365529
// MI455X (gfx1250) — compile-verified, matrix operand order checked by reading, not run on hardware
//
#include <hip/hip_runtime.h>
#include <hip/hip_bf16.h>
#include <math.h>

typedef __attribute__((ext_vector_type(16))) _Float16 v16h;
typedef __attribute__((ext_vector_type(8)))  float    v8f;

#define B_ 4
#define H_ 16
#define L_ 1024
#define ROWS_PER_TENSOR (B_ * H_ * L_)   // 65536

// ---------------------------------------------------------------------------
// Phase 1: per-row statistics. One wave (32 lanes) per row of 1024 floats.
// Each lane loads 8 coalesced float4 (global_load_b128), reduces locally,
// then a 5-step wave32 butterfly produces {mean, std(ddof=1), max, min}.
// Pure bandwidth pass: 512 MB read @ 23.3 TB/s -> ~22 us floor.
// ---------------------------------------------------------------------------
__global__ __launch_bounds__(256) void msg_stats_kernel(
    const float* __restrict__ a1, const float* __restrict__ a2,
    float4* __restrict__ statsOut)
{
    const int lane = threadIdx.x & 31;
    const int wave = threadIdx.x >> 5;
    const long gidx = (long)blockIdx.x * 8 + wave;      // 0 .. 131071
    const int  ten  = (int)(gidx >> 16);                // which tensor
    const long row  = gidx & (ROWS_PER_TENSOR - 1);

    const float4* src = (const float4*)((ten == 0 ? a1 : a2) + row * (long)L_);

    float s  = 0.f;
    float sq = 0.f;
    float mx = -3.402823466e38f;
    float mn =  3.402823466e38f;

#pragma unroll
    for (int i = 0; i < 8; ++i) {
        float4 v = src[i * 32 + lane];
        s  += v.x + v.y + v.z + v.w;
        sq += v.x * v.x + v.y * v.y + v.z * v.z + v.w * v.w;
        mx = fmaxf(mx, fmaxf(fmaxf(v.x, v.y), fmaxf(v.z, v.w)));
        mn = fminf(mn, fminf(fminf(v.x, v.y), fminf(v.z, v.w)));
    }
#pragma unroll
    for (int m = 16; m >= 1; m >>= 1) {
        s  += __shfl_xor(s,  m, 32);
        sq += __shfl_xor(sq, m, 32);
        mx  = fmaxf(mx, __shfl_xor(mx, m, 32));
        mn  = fminf(mn, __shfl_xor(mn, m, 32));
    }
    if (lane == 0) {
        const float n   = (float)L_;
        float mean = s / n;
        float var  = (sq - s * s / n) / (n - 1.0f);
        var = var > 0.f ? var : 0.f;
        float4 o;
        o.x = mean; o.y = sqrtf(var); o.z = mx; o.w = mn;
        statsOut[gidx] = o;
    }
}

// ---------------------------------------------------------------------------
// Phase 2: gate MLP + blend. One block per (b,h). 8 waves, each wave owns
// M=16-window tiles: 32 tiles (ws=2) + 16 (ws=4) + 8 (ws=8) = 56 tiles.
// Per tile: assemble g[16][32] (8 real features, zero pad) in per-wave LDS,
// gather A/B fragments per CDNA5 16-bit layouts, run:
//   layer1: 2x v_wmma_f32_16x16x32_f16  (N-tiles of the 32 hidden units)
//   layer2: 1x v_wmma_f32_16x16x32_f16  (32 -> 16)
//   layer3: 16->1 dot via 16-lane shfl_xor butterflies, sigmoid.
// Then blend the three per-window alpha arrays with softmax(scale_weights).
// ---------------------------------------------------------------------------
__global__ __launch_bounds__(256) void msg_gate_kernel(
    const float4* __restrict__ stats,
    const float* __restrict__ W1, const float* __restrict__ b1,
    const float* __restrict__ W2, const float* __restrict__ b2,
    const float* __restrict__ W3, const float* __restrict__ b3,
    const float* __restrict__ sw,
    float* __restrict__ out)
{
    const int bh   = blockIdx.x;        // 0..63
    const int tid  = threadIdx.x;
    const int lane = tid & 31;
    const int wave = tid >> 5;

    __shared__ _Float16 w1B[3][32][32];     // B-matrix staging, [k][n], k>=8 zero
    __shared__ _Float16 w2B[3][32][16];     // [k][n]
    __shared__ float    b1s[3][32];
    __shared__ float    b2s[3][16];
    __shared__ float    w3s[3][16];
    __shared__ float    b3s[3];
    __shared__ float    alphaS[896];        // 512 + 256 + 128
    __shared__ _Float16 stage[8][16][32];   // per-wave A-matrix staging

    // ---- stage weights (transposed to K-major, f16) ----
    for (int i = tid; i < 3 * 32 * 32; i += 256) {
        int s = i >> 10, k = (i >> 5) & 31, n = i & 31;
        w1B[s][k][n] = (k < 8) ? (_Float16)W1[(s * 32 + n) * 8 + k] : (_Float16)0.f;
    }
    for (int i = tid; i < 3 * 32 * 16; i += 256) {
        int s = i >> 9, k = (i >> 4) & 31, n = i & 15;
        w2B[s][k][n] = (_Float16)W2[(s * 16 + n) * 32 + k];
    }
    if (tid < 96) b1s[tid >> 5][tid & 31] = b1[tid];
    if (tid < 48) b2s[tid >> 4][tid & 15] = b2[tid];
    if (tid < 48) w3s[tid >> 4][tid & 15] = W3[tid];
    if (tid < 3)  b3s[tid] = b3[tid];
    __syncthreads();

    const int kb = (lane < 16) ? 0 : 8;     // K base for 16-bit A/B fragments

    for (int t = wave; t < 56; t += 8) {
        int s, ws, w0, aoff;
        if (t < 32)      { s = 0; ws = 2; w0 = t * 16;        aoff = 0;   }
        else if (t < 48) { s = 1; ws = 4; w0 = (t - 32) * 16; aoff = 512; }
        else             { s = 2; ws = 8; w0 = (t - 48) * 16; aoff = 768; }

        // ---- features into per-wave staging (zero pad K 8..31) ----
        {
            int row = lane & 15, cbase = (lane >> 4) * 16;
#pragma unroll
            for (int j = 0; j < 16; ++j) stage[wave][row][cbase + j] = (_Float16)0.f;
        }
        {
            int widx = lane & 15, ten = lane >> 4;  // lanes 0-15: attn_1, 16-31: attn_2
            long rbase = (long)ten * ROWS_PER_TENSOR + (long)bh * L_
                       + (long)(w0 + widx) * ws;
            float4 acc = {0.f, 0.f, 0.f, 0.f};
            for (int r = 0; r < ws; ++r) {
                float4 v = stats[rbase + r];
                acc.x += v.x; acc.y += v.y; acc.z += v.z; acc.w += v.w;
            }
            float inv = 1.0f / (float)ws;
            stage[wave][widx][ten * 4 + 0] = (_Float16)(acc.x * inv);
            stage[wave][widx][ten * 4 + 1] = (_Float16)(acc.y * inv);
            stage[wave][widx][ten * 4 + 2] = (_Float16)(acc.z * inv);
            stage[wave][widx][ten * 4 + 3] = (_Float16)(acc.w * inv);
        }

        // ---- gather A (16x32 f16 layout) and layer-1 B fragments ----
        v16h a, bA, bB;
        {
            int m = lane & 15, n = lane & 15;
#pragma unroll
            for (int v = 0; v < 8; ++v) {
                int k = ((v & 3) << 1) + ((v >> 2) << 4) + kb;
                a[2 * v]     = stage[wave][m][k];
                a[2 * v + 1] = stage[wave][m][k + 1];
                bA[2 * v]     = w1B[s][k][n];
                bA[2 * v + 1] = w1B[s][k + 1][n];
                bB[2 * v]     = w1B[s][k][n + 16];
                bB[2 * v + 1] = w1B[s][k + 1][n + 16];
            }
        }
        v8f c0 = {}, c1 = {};
        c0 = __builtin_amdgcn_wmma_f32_16x16x32_f16(false, a, false, bA, (short)0, c0, false, false);
        c1 = __builtin_amdgcn_wmma_f32_16x16x32_f16(false, a, false, bB, (short)0, c1, false, false);

        // ---- bias + ReLU, restage h[16][32] ----
        {
            int col = lane & 15, rofs = (lane < 16) ? 0 : 8;
            float bb0 = b1s[s][col], bb1 = b1s[s][col + 16];
#pragma unroll
            for (int r = 0; r < 8; ++r) {
                float h0 = c0[r] + bb0; h0 = h0 > 0.f ? h0 : 0.f;
                float h1 = c1[r] + bb1; h1 = h1 > 0.f ? h1 : 0.f;
                stage[wave][r + rofs][col]      = (_Float16)h0;
                stage[wave][r + rofs][col + 16] = (_Float16)h1;
            }
        }

        // ---- layer 2: h(16x32) @ W2^T(32x16) ----
        v16h a2, b2f;
        {
            int m = lane & 15, n = lane & 15;
#pragma unroll
            for (int v = 0; v < 8; ++v) {
                int k = ((v & 3) << 1) + ((v >> 2) << 4) + kb;
                a2[2 * v]     = stage[wave][m][k];
                a2[2 * v + 1] = stage[wave][m][k + 1];
                b2f[2 * v]     = w2B[s][k][n];
                b2f[2 * v + 1] = w2B[s][k + 1][n];
            }
        }
        v8f c2 = {};
        c2 = __builtin_amdgcn_wmma_f32_16x16x32_f16(false, a2, false, b2f, (short)0, c2, false, false);

        // ---- layer 3: bias + ReLU + 16->1 dot + sigmoid ----
        {
            int col = lane & 15;
            float w3c  = w3s[s][col];
            float bias = b2s[s][col];
            float b3v  = b3s[s];
#pragma unroll
            for (int r = 0; r < 8; ++r) {
                float h = c2[r] + bias; h = h > 0.f ? h : 0.f;
                float p = h * w3c;
                p += __shfl_xor(p, 8, 32);
                p += __shfl_xor(p, 4, 32);
                p += __shfl_xor(p, 2, 32);
                p += __shfl_xor(p, 1, 32);
                float alpha = 1.0f / (1.0f + __expf(-(p + b3v)));
                if ((lane & 15) == 0)
                    alphaS[aoff + w0 + r + ((lane < 16) ? 0 : 8)] = alpha;
            }
        }
    }
    __syncthreads();

    // ---- softmax(scale_weights) blend, write [bh, 0..1023] ----
    float sw0 = sw[0], sw1 = sw[1], sw2 = sw[2];
    float mw  = fmaxf(sw0, fmaxf(sw1, sw2));
    float e0 = __expf(sw0 - mw), e1 = __expf(sw1 - mw), e2 = __expf(sw2 - mw);
    float inv = 1.0f / (e0 + e1 + e2);
    e0 *= inv; e1 *= inv; e2 *= inv;
#pragma unroll
    for (int j = 0; j < 4; ++j) {
        int l = tid + j * 256;
        out[(long)bh * L_ + l] =
            e0 * alphaS[l >> 1] + e1 * alphaS[512 + (l >> 2)] + e2 * alphaS[768 + (l >> 3)];
    }
}

extern "C" void kernel_launch(void* const* d_in, const int* in_sizes, int n_in,
                              void* d_out, int out_size, void* d_ws, size_t ws_size,
                              hipStream_t stream) {
    const float* a1 = (const float*)d_in[0];
    const float* a2 = (const float*)d_in[1];
    const float* W1 = (const float*)d_in[2];
    const float* b1 = (const float*)d_in[3];
    const float* W2 = (const float*)d_in[4];
    const float* b2 = (const float*)d_in[5];
    const float* W3 = (const float*)d_in[6];
    const float* b3 = (const float*)d_in[7];
    const float* sw = (const float*)d_in[8];

    float4* statsBuf = (float4*)d_ws;   // 2 * 65536 float4 = 2 MB

    // 131072 rows total (2 tensors x 65536), 8 waves per block
    msg_stats_kernel<<<16384, 256, 0, stream>>>(a1, a2, statsBuf);
    msg_gate_kernel<<<B_ * H_, 256, 0, stream>>>(statsBuf, W1, b1, W2, b2, W3, b3, sw,
                                                 (float*)d_out);
}
